// GNNModel_9234179686641
// MI455X (gfx1250) — matrix operand order repaired and checked by reading, not compiled, not run on hardware
//
#include <hip/hip_runtime.h>
#include <hip/hip_bf16.h>
#include <cstdint>

#define N_NODES 1000000
#define F 128
#define NCLS 16

typedef __attribute__((ext_vector_type(16))) __bf16 v16bf;
typedef __attribute__((ext_vector_type(8)))  float  v8f;

// ---------------- index preprocessing ----------------

__global__ void k_init_fo(int* __restrict__ fo, int n) {
    int i = blockIdx.x * blockDim.x + threadIdx.x;
    if (i < n) fo[i] = 0x7fffffff;
}

__global__ void k_first_occ(const int* __restrict__ ei, int* __restrict__ fo, int n) {
    int i = blockIdx.x * blockDim.x + threadIdx.x;
    if (i < n) atomicMin(&fo[ei[2 * i]], i);
}

// 1024 positions per block (256 thr x 4): partial sums of first-occurrence flags
__global__ __launch_bounds__(256) void k_block_sum(const int* __restrict__ ei,
                                                   const int* __restrict__ fo,
                                                   int* __restrict__ bsum, int n) {
    __shared__ int red[256];
    int tid = threadIdx.x;
    int base = blockIdx.x * 1024 + tid * 4;
    int s = 0;
#pragma unroll
    for (int j = 0; j < 4; j++) {
        int p = base + j;
        if (p < n) s += (fo[ei[2 * p]] == p) ? 1 : 0;
    }
    red[tid] = s;
    __syncthreads();
    for (int st = 128; st > 0; st >>= 1) {
        if (tid < st) red[tid] += red[tid + st];
        __syncthreads();
    }
    if (tid == 0) bsum[blockIdx.x] = red[0];
}

// single-block inclusive scan of <=1024 block sums -> exclusive offsets
__global__ __launch_bounds__(1024) void k_scan_bsum(const int* __restrict__ bsum,
                                                    int* __restrict__ boff, int nb) {
    __shared__ int a[1024];
    int t = threadIdx.x;
    int v = (t < nb) ? bsum[t] : 0;
    a[t] = v;
    __syncthreads();
    for (int off = 1; off < 1024; off <<= 1) {
        int xv = (t >= off) ? a[t - off] : 0;
        __syncthreads();
        a[t] += xv;
        __syncthreads();
    }
    if (t < nb) boff[t] = a[t] - v;  // exclusive
}

// per-position exclusive scan of flags: S[p] = #first-occurrences at positions < p
__global__ __launch_bounds__(256) void k_scan_pos(const int* __restrict__ ei,
                                                  const int* __restrict__ fo,
                                                  const int* __restrict__ boff,
                                                  int* __restrict__ S, int n) {
    __shared__ int a[256];
    int tid = threadIdx.x;
    int base = blockIdx.x * 1024 + tid * 4;
    int f[4];
    int s = 0;
#pragma unroll
    for (int j = 0; j < 4; j++) {
        int p = base + j;
        f[j] = (p < n && fo[ei[2 * p]] == p) ? 1 : 0;
        s += f[j];
    }
    a[tid] = s;
    __syncthreads();
    for (int off = 1; off < 256; off <<= 1) {
        int xv = (tid >= off) ? a[tid - off] : 0;
        __syncthreads();
        a[tid] += xv;
        __syncthreads();
    }
    int run = boff[blockIdx.x] + (a[tid] - s);
#pragma unroll
    for (int j = 0; j < 4; j++) {
        int p = base + j;
        if (p < n) { S[p] = run; run += f[j]; }
    }
}

// mapped[i] = rank[seg[seg[i]]] = S[first_occ[seg[seg[i]]]]
__global__ void k_mapped(const int* __restrict__ ei, const int* __restrict__ fo,
                         const int* __restrict__ S, int* __restrict__ mapped, int n) {
    int i = blockIdx.x * blockDim.x + threadIdx.x;
    if (i < n) {
        int s1 = ei[2 * i];
        int s2 = ei[2 * s1];
        mapped[i] = S[fo[s2]];
    }
}

__global__ void k_zero4(float4* __restrict__ p, size_t n4) {
    size_t i = (size_t)blockIdx.x * blockDim.x + threadIdx.x;
    if (i < n4) p[i] = make_float4(0.f, 0.f, 0.f, 0.f);
}

// ---------------- fused GEMM1 (bf16 WMMA) + bias + ReLU + atomic scatter ----------------
// block = 256 thr = 8 waves; wave computes 16 rows x 128 cols, scatters into agg[mapped[row]]

__global__ __launch_bounds__(256) void k_gemm1_scatter(const float* __restrict__ x,
                                                       const float* __restrict__ W1,
                                                       const float* __restrict__ b1,
                                                       const int* __restrict__ mapped,
                                                       float* __restrict__ agg) {
    __shared__ __attribute__((aligned(32))) __bf16 w1t[F * F];  // transposed: [n][k]
    __shared__ float b1s[F];
    int tid = threadIdx.x;
    for (int idx = tid; idx < F * F; idx += 256) {
        int k = idx >> 7, n = idx & 127;
        w1t[n * F + k] = (__bf16)W1[idx];
    }
    if (tid < F) b1s[tid] = b1[tid];
    __syncthreads();

    int wave = tid >> 5, lane = tid & 31, hi = lane >> 4, lo = lane & 15;
    int rowBase = blockIdx.x * 128 + wave * 16;
    if (rowBase >= N_NODES) return;  // wave-uniform; N divisible by 16

    const float* xrow = x + (size_t)(rowBase + lo) * F;

    v8f acc[8] = {};
#pragma unroll
    for (int kc = 0; kc < 4; kc++) {
        int kb = kc * 32;
        v16bf a;
        const float* p0 = xrow + kb + hi * 16;
        const float* p1 = xrow + kb + hi * 16 + 8;
#pragma unroll
        for (int j = 0; j < 8; j++) {
            a[j]     = (__bf16)p0[j];
            a[j + 8] = (__bf16)p1[j];
        }
#pragma unroll
        for (int nt = 0; nt < 8; nt++) {
            const v16bf b = *(const v16bf*)&w1t[(lo + nt * 16) * F + kb + hi * 16];
            acc[nt] = __builtin_amdgcn_wmma_f32_16x16x32_bf16(
                false, a, false, b, (short)0, acc[nt], false, false);
        }
    }

    int dstRow[8];
#pragma unroll
    for (int r = 0; r < 8; r++) dstRow[r] = mapped[rowBase + hi * 8 + r];

#pragma unroll
    for (int nt = 0; nt < 8; nt++) {
        int n = lo + nt * 16;
        float bias = b1s[n];
#pragma unroll
        for (int r = 0; r < 8; r++) {
            float v = acc[nt][r] + bias;
            v = v > 0.f ? v : 0.f;  // ReLU before segment-sum
            atomicAdd(&agg[(size_t)dstRow[r] * F + n], v);
        }
    }
}

// ---------------- GEMM2: out = relu(agg) @ W2 + b2 (bf16 WMMA) ----------------

__global__ __launch_bounds__(256) void k_gemm2(const float* __restrict__ agg,
                                               const float* __restrict__ W2,
                                               const float* __restrict__ b2,
                                               float* __restrict__ out) {
    __shared__ __attribute__((aligned(32))) __bf16 w2t[NCLS * F];  // transposed: [c][k]
    __shared__ float b2s[NCLS];
    int tid = threadIdx.x;
    for (int idx = tid; idx < F * NCLS; idx += 256) {
        int k = idx >> 4, c = idx & 15;
        w2t[c * F + k] = (__bf16)W2[idx];
    }
    if (tid < NCLS) b2s[tid] = b2[tid];
    __syncthreads();

    int wave = tid >> 5, lane = tid & 31, hi = lane >> 4, lo = lane & 15;
    int rowBase = blockIdx.x * 128 + wave * 16;
    if (rowBase >= N_NODES) return;

    const float* arow = agg + (size_t)(rowBase + lo) * F;

    v8f acc = {};
#pragma unroll
    for (int kc = 0; kc < 4; kc++) {
        int kb = kc * 32;
        v16bf a;
        const float* p0 = arow + kb + hi * 16;
        const float* p1 = arow + kb + hi * 16 + 8;
#pragma unroll
        for (int j = 0; j < 8; j++) {
            float u0 = p0[j]; u0 = u0 > 0.f ? u0 : 0.f;  // ReLU(agg)
            float u1 = p1[j]; u1 = u1 > 0.f ? u1 : 0.f;
            a[j]     = (__bf16)u0;
            a[j + 8] = (__bf16)u1;
        }
        const v16bf b = *(const v16bf*)&w2t[lo * F + kb + hi * 16];
        acc = __builtin_amdgcn_wmma_f32_16x16x32_bf16(
            false, a, false, b, (short)0, acc, false, false);
    }

    float bias = b2s[lo];
#pragma unroll
    for (int r = 0; r < 8; r++) {
        int m = rowBase + r + 8 * hi;
        out[(size_t)m * NCLS + lo] = acc[r] + bias;
    }
}

// ---------------- launcher ----------------

extern "C" void kernel_launch(void* const* d_in, const int* in_sizes, int n_in,
                              void* d_out, int out_size, void* d_ws, size_t ws_size,
                              hipStream_t stream) {
    const float* x  = (const float*)d_in[0];
    const int*   ei = (const int*)d_in[1];
    const float* W1 = (const float*)d_in[2];
    const float* b1 = (const float*)d_in[3];
    const float* W2 = (const float*)d_in[4];
    const float* b2 = (const float*)d_in[5];
    float* out = (float*)d_out;

    const int n = N_NODES;
    char* ws = (char*)d_ws;
    size_t off = 0;
    auto alloc = [&](size_t bytes) {
        void* p = ws + off;
        off = (off + bytes + 255) & ~(size_t)255;
        return p;
    };
    float* agg   = (float*)alloc((size_t)n * F * sizeof(float));  // 512 MB
    int*   fo    = (int*)alloc((size_t)n * sizeof(int));
    int*   S     = (int*)alloc((size_t)n * sizeof(int));
    int*   mapped= (int*)alloc((size_t)n * sizeof(int));
    int*   bsum  = (int*)alloc(1024 * sizeof(int));
    int*   boff  = (int*)alloc(1024 * sizeof(int));

    const int nb1024 = (n + 1023) / 1024;      // 977 <= 1024
    const int nb256  = (n + 255) / 256;

    k_init_fo<<<nb256, 256, 0, stream>>>(fo, n);
    k_first_occ<<<nb256, 256, 0, stream>>>(ei, fo, n);
    k_block_sum<<<nb1024, 256, 0, stream>>>(ei, fo, bsum, n);
    k_scan_bsum<<<1, 1024, 0, stream>>>(bsum, boff, nb1024);
    k_scan_pos<<<nb1024, 256, 0, stream>>>(ei, fo, boff, S, n);
    k_mapped<<<nb256, 256, 0, stream>>>(ei, fo, S, mapped, n);

    size_t n4 = (size_t)n * F / 4;  // 32M float4
    k_zero4<<<(unsigned)((n4 + 255) / 256), 256, 0, stream>>>((float4*)agg, n4);

    const int gemmBlocks = (n + 127) / 128;  // 7813, 8 waves x 16 rows each
    k_gemm1_scatter<<<gemmBlocks, 256, 0, stream>>>(x, W1, b1, mapped, agg);
    k_gemm2<<<gemmBlocks, 256, 0, stream>>>(agg, W2, b2, out);
}
